// MultiHeadSelfAttention_26766236188907
// MI455X (gfx1250) — matrix operand order repaired and checked by reading, not compiled, not run on hardware
//
#include <hip/hip_runtime.h>
#include <hip/hip_bf16.h>

// ---------------------------------------------------------------------------
// MultiHeadSelfAttention for MI455X (gfx1250, wave32, WMMA bf16)
// B=4, S=2048, D=1024, H=16, d_k=64
// Data movement: TDM (tensor_load_to_lds) + async global->LDS + VALU transpose
// ---------------------------------------------------------------------------

#define B_   4
#define S_   2048
#define D_   1024
#define H_   16
#define DK_  64
#define MTOT (B_ * S_)          // 8192 rows

typedef __attribute__((ext_vector_type(16))) __bf16    v16bf;
typedef __attribute__((ext_vector_type(8)))  float     v8f;
typedef __attribute__((ext_vector_type(2)))  __bf16    v2bf;
typedef __attribute__((ext_vector_type(4)))  unsigned  uint4v;
typedef __attribute__((ext_vector_type(8)))  int       int8v;
typedef __attribute__((ext_vector_type(4)))  int       int4v;

union Frag16 {                  // 16 bf16 = 32 B (A or B operand of 16x16x32)
    v16bf v;
    uint4 q[2];
    uint2 d[4];
    unsigned short u[16];
};
union FragC {                   // 8 f32 accumulator (16x16 C/D tile)
    v8f   v;
    float f[8];
};

__device__ __forceinline__ unsigned short f2bf(float x) {
    unsigned u = __builtin_bit_cast(unsigned, x);
    unsigned r = u + 0x7FFFu + ((u >> 16) & 1u);   // round-to-nearest-even
    return (unsigned short)(r >> 16);
}

// pack two f32 -> packed bf16x2 (one VALU op when v_cvt_pk_bf16_f32 exists)
__device__ __forceinline__ unsigned pack_bf16(float a, float b) {
#if __has_builtin(__builtin_amdgcn_cvt_pk_bf16_f32)
    v2bf r = __builtin_amdgcn_cvt_pk_bf16_f32(a, b);
    return __builtin_bit_cast(unsigned, r);
#else
    return (unsigned)f2bf(a) | ((unsigned)f2bf(b) << 16);
#endif
}

// ---------------------------------------------------------------------------
// Async HBM -> LDS DMA (per-lane 16B), tracked by ASYNCcnt
// ---------------------------------------------------------------------------
__device__ __forceinline__ void async_load_b128(void* lds_dst, const void* gsrc) {
    unsigned lds_off = (unsigned)(size_t)lds_dst;   // low 32 bits = LDS offset
    asm volatile("global_load_async_to_lds_b128 %0, %1, off"
                 :: "v"(lds_off), "v"(gsrc) : "memory");
}
__device__ __forceinline__ void wait_async0() {
    asm volatile("s_wait_asynccnt 0x0" ::: "memory");
}

// ---------------------------------------------------------------------------
// Tensor Data Mover: 2D tile HBM -> LDS, bf16 elements, optional LDS padding.
// D# built per CDNA5 ISA ch.8: group0 {count=1, lds_addr, global_addr, type=2},
// group1 {data_size=1(2B), pad fields, tensor/tile dims, dim0 stride}.
// Tracked by TENSORcnt. Issue from ONE wave only.
// ---------------------------------------------------------------------------
__device__ __forceinline__ void tdm_load_2d(unsigned lds_addr, const void* gptr,
                                            unsigned tile_d0, unsigned tile_d1,
                                            unsigned long long stride0_elems,
                                            unsigned pad_interval_code,
                                            unsigned pad_amount_code,
                                            int pad_en) {
    unsigned long long ga = (unsigned long long)(size_t)gptr;
    uint4v g0;
    g0[0] = 1u;                                   // count=1 (valid user D#)
    g0[1] = lds_addr;                             // LDS byte address
    g0[2] = (unsigned)ga;                         // global addr [31:0]
    g0[3] = (unsigned)((ga >> 32) & 0x01FFFFFFull) | (2u << 30); // [56:32]|type=2

    int8v g1;
    unsigned w0 = (1u << 16)                      // data_size = 1 -> 2 bytes
                | ((unsigned)(pad_en ? 1u : 0u) << 20)
                | (pad_interval_code << 22)
                | (pad_amount_code << 25);
    unsigned td0 = tile_d0;                       // tensor dims = tile dims
    unsigned td1 = tile_d1;                       // (tile fully in-bounds)
    unsigned long long s0 = stride0_elems;
    g1[0] = (int)w0;
    g1[1] = (int)((td0 & 0xFFFFu) << 16);                     // tensor_dim0.lo
    g1[2] = (int)((td0 >> 16) | ((td1 & 0xFFFFu) << 16));     // .hi | dim1.lo
    g1[3] = (int)((td1 >> 16) | (tile_d0 << 16));             // dim1.hi | tile0
    g1[4] = (int)(tile_d1 & 0xFFFFu);                         // tile1 | tile2=0
    g1[5] = (int)(s0 & 0xFFFFFFFFull);                        // stride0[31:0]
    g1[6] = (int)((s0 >> 32) & 0xFFFFull);                    // stride0[47:32]
    g1[7] = 0;                                                // stride1=0 (2D)

    int4v z4 = {0, 0, 0, 0};
#if __clang_major__ >= 23
    int8v z8 = {0, 0, 0, 0, 0, 0, 0, 0};
    __builtin_amdgcn_tensor_load_to_lds(g0, g1, z4, z4, z8, 0);
#else
    __builtin_amdgcn_tensor_load_to_lds(g0, g1, z4, z4, 0);
#endif
}

// ---------------------------------------------------------------------------
// f32 -> bf16 downconvert (vectorized)
// ---------------------------------------------------------------------------
__global__ void cvt_f32_bf16(const float* __restrict__ in,
                             unsigned short* __restrict__ out, int n) {
    int i = (blockIdx.x * blockDim.x + threadIdx.x) * 4;
    if (i + 3 < n) {
        float4 f = *(const float4*)(in + i);
        uint2 p;
        p.x = pack_bf16(f.x, f.y);
        p.y = pack_bf16(f.z, f.w);
        *(uint2*)(out + i) = p;
    }
}

// ---------------------------------------------------------------------------
// GEMM + bias: out(MxN) = A(MxK,bf16) @ W(KxN,bf16) + bias
// Block: 256 threads (8 waves), tile 128x64, K-step 32.
// A tile staged by TDM (one descriptor, padded rows -> LDA=40 halfwords);
// W tile transposed via VALU.
// ---------------------------------------------------------------------------
#define GM  128
#define GN  64
#define GK  32
#define LDA 40   // padded stride (halfwords) == TDM pad: 64B data + 16B pad
#define LDB 40

template <bool OUT_BF16>
__global__ void gemm_bias(const unsigned short* __restrict__ A,
                          const unsigned short* __restrict__ W,
                          const float* __restrict__ bias,
                          void* __restrict__ out,
                          int M, int N, int K) {
    __shared__ unsigned short As[GM * LDA];   // 10.0 KB
    __shared__ unsigned short Bs[GN * LDB];   //  5.0 KB

    const int tid  = threadIdx.x;
    const int lane = tid & 31;
    const int wave = tid >> 5;        // 0..7
    const int wm   = wave >> 1;       // 0..3 (M direction, 32 rows each)
    const int wn   = wave & 1;        // 0..1 (N direction, 32 cols each)
    const int m0   = blockIdx.y * GM;
    const int n0   = blockIdx.x * GN;

    FragC acc[2][2];
#pragma unroll
    for (int i = 0; i < 2; ++i)
#pragma unroll
        for (int j = 0; j < 2; ++j)
#pragma unroll
            for (int e = 0; e < 8; ++e) acc[i][j].f[e] = 0.0f;

    for (int k0 = 0; k0 < K; k0 += GK) {
        __syncthreads();
        // ---- stage A tile via TDM: 128 rows x 32 k, rows padded 64B+16B ----
        if (tid < 32) {   // single wave issues the DMA descriptor
            tdm_load_2d((unsigned)(size_t)&As[0],
                        A + (size_t)m0 * K + k0,
                        /*tile_d0=*/GK, /*tile_d1=*/GM,
                        /*stride0=*/(unsigned long long)K,
                        /*pad_interval(64B)=*/3u, /*pad_amount(16B)=*/3u, 1);
            __builtin_amdgcn_s_wait_tensorcnt(0);
        }
        // ---- stage W tile transposed: Bs[n][k], n0..n0+63, k0..k0+31 ----
        {
            int k  = tid >> 3;                // 0..31
            int nc = (tid & 7) * 8;           // 0..56
            const unsigned short* src = W + (size_t)(k0 + k) * N + n0 + nc;
            uint4 vld = *(const uint4*)src;
            unsigned short tmp[8];
            *(uint4*)tmp = vld;
#pragma unroll
            for (int j = 0; j < 8; ++j) Bs[(nc + j) * LDB + k] = tmp[j];
            if (k0 + GK < K)                  // prefetch next W panel
                __builtin_prefetch(src + (size_t)GK * N, 0, 1);
        }
        __syncthreads();

        // ---- load fragments ----
        Frag16 af[2], bf[2];
        const int arow = lane & 15;
        const int akb  = (lane >> 4) * 8;     // A: lanes16-31 hold K+8
        const int bkb  = (lane >> 4) * 8;     // B: lanes16-31 hold K+8
#pragma unroll
        for (int i = 0; i < 2; ++i) {
            int r = wm * 32 + i * 16 + arow;
            af[i].d[0] = *(const uint2*)(&As[r * LDA + akb + 0]);
            af[i].d[1] = *(const uint2*)(&As[r * LDA + akb + 4]);
            af[i].d[2] = *(const uint2*)(&As[r * LDA + akb + 16]);
            af[i].d[3] = *(const uint2*)(&As[r * LDA + akb + 20]);
            int nn = wn * 32 + i * 16 + (lane & 15);
            bf[i].d[0] = *(const uint2*)(&Bs[nn * LDB + bkb + 0]);
            bf[i].d[1] = *(const uint2*)(&Bs[nn * LDB + bkb + 4]);
            bf[i].d[2] = *(const uint2*)(&Bs[nn * LDB + bkb + 16]);
            bf[i].d[3] = *(const uint2*)(&Bs[nn * LDB + bkb + 20]);
        }
#pragma unroll
        for (int i = 0; i < 2; ++i)
#pragma unroll
            for (int j = 0; j < 2; ++j)
                acc[i][j].v = __builtin_amdgcn_wmma_f32_16x16x32_bf16(
                    false, af[i].v, false, bf[j].v, (short)0, acc[i][j].v,
                    false, false);
    }

    // ---- epilogue: bias add + store (C layout: col=lane&15, rows=(lane>>4)*8+e)
    const int col   = lane & 15;
    const int rbase = (lane >> 4) * 8;
#pragma unroll
    for (int i = 0; i < 2; ++i)
#pragma unroll
        for (int j = 0; j < 2; ++j) {
            int n = n0 + wn * 32 + j * 16 + col;
            float bv = bias[n];
            if (OUT_BF16) {
#pragma unroll
                for (int e = 0; e < 8; e += 2) {
                    int m = m0 + wm * 32 + i * 16 + rbase + e;
                    unsigned pk = pack_bf16(acc[i][j].f[e] + bv,
                                            acc[i][j].f[e + 1] + bv);
                    ((unsigned short*)out)[(size_t)m * N + n] =
                        (unsigned short)pk;
                    ((unsigned short*)out)[(size_t)(m + 1) * N + n] =
                        (unsigned short)(pk >> 16);
                }
            } else {
#pragma unroll
                for (int e = 0; e < 8; ++e) {
                    int m = m0 + wm * 32 + i * 16 + rbase + e;
                    ((float*)out)[(size_t)m * N + n] = acc[i][j].f[e] + bv;
                }
            }
        }
}

// ---------------------------------------------------------------------------
// Flash attention per (b,h): block = 128 threads (4 waves), 64 Q rows/block,
// stream over 64-key tiles with online softmax. All matmuls on WMMA bf16.
// K tile staged with async HBM->LDS DMA; V tile transposed via VALU.
// ---------------------------------------------------------------------------
#define ALD 72   // padded LDS stride (halfwords), 144 B (16B-aligned rows)

__global__ void attn_kernel(const unsigned short* __restrict__ Q,
                            const unsigned short* __restrict__ K,
                            const unsigned short* __restrict__ V,
                            unsigned short* __restrict__ O) {
    __shared__ unsigned short Ks[64 * ALD];       // K tile, row-major
    __shared__ unsigned short Vt[64 * ALD];       // V tile, transposed [d][key]
    __shared__ unsigned short Ps[4 * 16 * ALD];   // per-wave P strip

    const int tid  = threadIdx.x;      // 0..127
    const int lane = tid & 31;
    const int wave = tid >> 5;         // 0..3
    const int bh   = blockIdx.y;       // 0..63
    const int b    = bh >> 4;
    const int h    = bh & 15;
    const int q0   = blockIdx.x * 64 + wave * 16;
    const size_t base = ((size_t)b * S_) * D_ + (size_t)h * DK_;
    const float scale = 0.125f;                 // 1/sqrt(64)
    const float log2e = 1.4426950408889634f;

    // ---- Q A-fragments (d_k = 64 -> two 16x32 fragments), straight from HBM
    Frag16 qf[2];
    {
        int row = q0 + (lane & 15);
        int kb  = (lane >> 4) * 8;
        const unsigned short* qp = Q + base + (size_t)row * D_;
#pragma unroll
        for (int ks = 0; ks < 2; ++ks) {
            qf[ks].q[0] = *(const uint4*)(qp + ks * 32 + kb);
            qf[ks].q[1] = *(const uint4*)(qp + ks * 32 + kb + 16);
        }
    }

    FragC o[4];
    float mrow[8], lrow[8];
#pragma unroll
    for (int dt = 0; dt < 4; ++dt)
#pragma unroll
        for (int e = 0; e < 8; ++e) o[dt].f[e] = 0.0f;
#pragma unroll
    for (int e = 0; e < 8; ++e) { mrow[e] = -1e30f; lrow[e] = 0.0f; }

    for (int kt = 0; kt < S_ / 64; ++kt) {
        __syncthreads();
        // ---- stage K (async DMA, row-major) and V (VALU transpose) ----
#pragma unroll
        for (int p = 0; p < 4; ++p) {
            int idx = tid + p * 128;        // 0..511
            int r   = idx >> 3;             // 0..63 (key row)
            int c   = (idx & 7) * 8;        // 0..56 (d chunk)
            size_t g = base + (size_t)(kt * 64 + r) * D_ + c;
            async_load_b128(&Ks[r * ALD + c], K + g);
            uint4 vv = *(const uint4*)(V + g);
            unsigned short tmp[8];
            *(uint4*)tmp = vv;
#pragma unroll
            for (int j = 0; j < 8; ++j) Vt[(c + j) * ALD + r] = tmp[j];
        }
        wait_async0();
        __syncthreads();

        // ---- S = Q @ K^T : 16 q-rows x 64 keys (8 WMMAs) ----
        FragC sc[4];
#pragma unroll
        for (int nt = 0; nt < 4; ++nt) {
#pragma unroll
            for (int e = 0; e < 8; ++e) sc[nt].f[e] = 0.0f;
#pragma unroll
            for (int ks = 0; ks < 2; ++ks) {
                Frag16 kf;
                int key = nt * 16 + (lane & 15);
                int db  = ks * 32 + (lane >> 4) * 8;
                const unsigned short* kp = &Ks[key * ALD + db];
                kf.q[0] = *(const uint4*)(kp);
                kf.q[1] = *(const uint4*)(kp + 16);
                sc[nt].v = __builtin_amdgcn_wmma_f32_16x16x32_bf16(
                    false, qf[ks].v, false, kf.v, (short)0, sc[nt].v,
                    false, false);
            }
        }

        // ---- online softmax (rows live in element index; cols in lanes) ----
#pragma unroll
        for (int e = 0; e < 8; ++e) {
            float mx = fmaxf(fmaxf(sc[0].f[e], sc[1].f[e]),
                             fmaxf(sc[2].f[e], sc[3].f[e])) * scale;
#pragma unroll
            for (int off = 1; off < 16; off <<= 1)
                mx = fmaxf(mx, __shfl_xor(mx, off, 32));
            float mnew = fmaxf(mrow[e], mx);
            float corr = exp2f((mrow[e] - mnew) * log2e);
            mrow[e] = mnew;
            float psum = 0.0f;
#pragma unroll
            for (int nt = 0; nt < 4; ++nt) {
                float p = exp2f((sc[nt].f[e] * scale - mnew) * log2e);
                sc[nt].f[e] = p;
                psum += p;
            }
#pragma unroll
            for (int off = 1; off < 16; off <<= 1)
                psum += __shfl_xor(psum, off, 32);
            lrow[e] = lrow[e] * corr + psum;
#pragma unroll
            for (int dt = 0; dt < 4; ++dt) o[dt].f[e] *= corr;
        }

        // ---- P: C-layout -> A-layout via per-wave LDS strip ----
        {
            int rb = (lane >> 4) * 8;
            int cl = lane & 15;
            unsigned short* pw = &Ps[wave * 16 * ALD];
#pragma unroll
            for (int nt = 0; nt < 4; ++nt)
#pragma unroll
                for (int e = 0; e < 8; e += 2) {
                    unsigned pk = pack_bf16(sc[nt].f[e], sc[nt].f[e + 1]);
                    pw[(rb + e) * ALD + nt * 16 + cl] = (unsigned short)pk;
                    pw[(rb + e + 1) * ALD + nt * 16 + cl] =
                        (unsigned short)(pk >> 16);
                }
        }
        asm volatile("s_wait_dscnt 0x0" ::: "memory");  // per-wave LDS RAW

        Frag16 pf[2];
        {
            const unsigned short* pr = &Ps[wave * 16 * ALD];
            int row = lane & 15;
            int kb  = (lane >> 4) * 8;
#pragma unroll
            for (int ks = 0; ks < 2; ++ks) {
                pf[ks].q[0] = *(const uint4*)(pr + row * ALD + ks * 32 + kb);
                pf[ks].q[1] = *(const uint4*)(pr + row * ALD + ks * 32 + kb + 16);
            }
        }
        // ---- O += P @ V (8 WMMAs) ----
#pragma unroll
        for (int dt = 0; dt < 4; ++dt) {
#pragma unroll
            for (int ks = 0; ks < 2; ++ks) {
                Frag16 vf;
                int d   = dt * 16 + (lane & 15);
                int kb2 = ks * 32 + (lane >> 4) * 8;
                const unsigned short* vp = &Vt[d * ALD + kb2];
                vf.q[0] = *(const uint4*)(vp);
                vf.q[1] = *(const uint4*)(vp + 16);
                o[dt].v = __builtin_amdgcn_wmma_f32_16x16x32_bf16(
                    false, pf[ks].v, false, vf.v, (short)0, o[dt].v,
                    false, false);
            }
        }
    }

    // ---- normalize + store context (bf16) ----
    {
        int cl = lane & 15;
        int rb = (lane >> 4) * 8;
#pragma unroll
        for (int dt = 0; dt < 4; ++dt) {
#pragma unroll
            for (int e = 0; e < 8; e += 2) {
                unsigned pk = pack_bf16(o[dt].f[e] / lrow[e],
                                        o[dt].f[e + 1] / lrow[e + 1]);
                int row = q0 + rb + e;
                O[base + (size_t)row * D_ + dt * 16 + cl] = (unsigned short)pk;
                O[base + (size_t)(row + 1) * D_ + dt * 16 + cl] =
                    (unsigned short)(pk >> 16);
            }
        }
    }
}

// ---------------------------------------------------------------------------
// Host-side launch
// ---------------------------------------------------------------------------
extern "C" void kernel_launch(void* const* d_in, const int* in_sizes, int n_in,
                              void* d_out, int out_size, void* d_ws,
                              size_t ws_size, hipStream_t stream) {
    const float* x  = (const float*)d_in[0];
    const float* Wq = (const float*)d_in[1];
    const float* bq = (const float*)d_in[2];
    const float* Wk = (const float*)d_in[3];
    const float* bk = (const float*)d_in[4];
    const float* Wv = (const float*)d_in[5];
    const float* bv = (const float*)d_in[6];
    const float* Wo = (const float*)d_in[7];
    const float* bo = (const float*)d_in[8];

    char* ws = (char*)d_ws;
    const size_t XB   = (size_t)MTOT * D_ * 2;   // 16 MiB bf16 activations
    const size_t WB   = (size_t)D_ * D_ * 2;     //  2 MiB bf16 weights
    unsigned short* xb  = (unsigned short*)(ws);
    unsigned short* wqb = (unsigned short*)(ws + XB);
    unsigned short* wkb = (unsigned short*)(ws + XB + 1 * WB);
    unsigned short* wvb = (unsigned short*)(ws + XB + 2 * WB);
    unsigned short* wob = (unsigned short*)(ws + XB + 3 * WB);
    unsigned short* Qb  = (unsigned short*)(ws + XB + 4 * WB);
    unsigned short* Kb  = (unsigned short*)(ws + 2 * XB + 4 * WB);
    unsigned short* Vb  = (unsigned short*)(ws + 3 * XB + 4 * WB);
    unsigned short* Cb  = (unsigned short*)(ws + 4 * XB + 4 * WB);

    // 1) downconvert activations + weights to bf16
    {
        int n = MTOT * D_;
        cvt_f32_bf16<<<n / 1024, 256, 0, stream>>>(x, xb, n);
        int nw = D_ * D_;
        cvt_f32_bf16<<<nw / 1024, 256, 0, stream>>>(Wq, wqb, nw);
        cvt_f32_bf16<<<nw / 1024, 256, 0, stream>>>(Wk, wkb, nw);
        cvt_f32_bf16<<<nw / 1024, 256, 0, stream>>>(Wv, wvb, nw);
        cvt_f32_bf16<<<nw / 1024, 256, 0, stream>>>(Wo, wob, nw);
    }

    // 2) Q/K/V projections (bf16 out)
    dim3 ggrid(D_ / GN, MTOT / GM);   // (16, 64)
    gemm_bias<true><<<ggrid, 256, 0, stream>>>(xb, wqb, bq, Qb, MTOT, D_, D_);
    gemm_bias<true><<<ggrid, 256, 0, stream>>>(xb, wkb, bk, Kb, MTOT, D_, D_);
    gemm_bias<true><<<ggrid, 256, 0, stream>>>(xb, wvb, bv, Vb, MTOT, D_, D_);

    // 3) flash attention: grid (S/64 q-tiles, B*H)
    attn_kernel<<<dim3(S_ / 64, B_ * H_), 128, 0, stream>>>(Qb, Kb, Vb, Cb);

    // 4) output projection (f32 out)
    gemm_bias<false><<<ggrid, 256, 0, stream>>>(Cb, wob, bo, (float*)d_out,
                                                MTOT, D_, D_);
}
